// MoEExperts_7155415515161
// MI455X (gfx1250) — matrix operand order repaired and checked by reading, not compiled, not run on hardware
//
#include <hip/hip_runtime.h>
#include <stdint.h>

// ---------------------------------------------------------------------------
// Grouped MoE SwiGLU FFN for gfx1250 (MI455X), bf16 WMMA path.
//   hiddens [4096,1024] f32, w1/w3 [8,1024,2816] f32, w2 [8,2816,1024] f32
//   out [4096,1024] f32.  Tokens contiguous per expert, 512 each.
// Phase 1: act = silu(x@w1) * (x@w3)  -> bf16 workspace
// Phase 2: out = act @ w2             -> f32
// ---------------------------------------------------------------------------

#define NUM_EXPERTS 8
#define HIDDEN      1024
#define INTER       2816
#define TOK_PER_E   512

typedef __attribute__((ext_vector_type(16))) __bf16 bf16x16;
typedef __attribute__((ext_vector_type(8)))  float  v8f;

union Frag {
    bf16x16 v;
    uint4   q[2];
};

__device__ __forceinline__ unsigned short f2bf(float f) {
    unsigned int u = __float_as_uint(f);
    u += 0x7FFFu + ((u >> 16) & 1u);      // round-to-nearest-even
    return (unsigned short)(u >> 16);
}

// Raw LDS byte offset of a generic pointer known to live in LDS.
typedef __attribute__((address_space(3))) const unsigned short* lds_u16_cptr;
__device__ __forceinline__ unsigned lds_off_u16(const unsigned short* p) {
    return (unsigned)(unsigned long long)(lds_u16_cptr)p;
}

__device__ __forceinline__ void wait_asynccnt0() {
#if __has_builtin(__builtin_amdgcn_s_wait_asynccnt)
    __builtin_amdgcn_s_wait_asynccnt(0);
#else
    asm volatile("s_wait_asynccnt 0x0" ::: "memory");
#endif
}

// LDS row stride in u16 elements: 32 payload + 8 pad, keeps 16B alignment.
#define LDA 40

// ---------------------------------------------------------------------------
// Kernel 1: act = silu(x@w1) * (x@w3), stored bf16 to workspace.
// Block: 256 threads (8 wave32s). WG tile 128(M) x 64(N), wave tile 32x32.
// ---------------------------------------------------------------------------
__global__ __launch_bounds__(256)
void moe_up_gate(const float* __restrict__ x,
                 const float* __restrict__ w1,
                 const float* __restrict__ w3,
                 unsigned short* __restrict__ act) {
    __shared__ unsigned short As [128 * LDA];
    __shared__ unsigned short B1s[ 64 * LDA];
    __shared__ unsigned short B3s[ 64 * LDA];

    const int e    = blockIdx.z;
    const int m0   = blockIdx.y * 128;     // token offset within expert
    const int n0   = blockIdx.x * 64;      // column offset within INTER
    const int tid  = threadIdx.x;
    const int lane = tid & 31;
    const int wid  = tid >> 5;
    const int wm   = wid & 3;              // 4 waves along M
    const int wn   = wid >> 2;             // 2 waves along N
    const int half = lane >> 4;
    const int l16  = lane & 15;

    const size_t tokBase = (size_t)e * TOK_PER_E + m0;
    const float* xg  = x  + tokBase * HIDDEN;
    const float* w1g = w1 + (size_t)e * HIDDEN * INTER + n0;
    const float* w3g = w3 + (size_t)e * HIDDEN * INTER + n0;

    v8f acc1[2][2] = {};
    v8f acc3[2][2] = {};

    for (int k0 = 0; k0 < HIDDEN; k0 += 32) {
        // ---- stage A: 128x32 f32 -> bf16 LDS [m][k]
#pragma unroll
        for (int j = 0; j < 4; ++j) {
            int f   = tid + 256 * j;               // float4 id (8 per row)
            int row = f >> 3;
            int kq  = (f & 7) << 2;
            float4 v = *(const float4*)(xg + (size_t)row * HIDDEN + k0 + kq);
            uint2 p;
            p.x = (unsigned int)f2bf(v.x) | ((unsigned int)f2bf(v.y) << 16);
            p.y = (unsigned int)f2bf(v.z) | ((unsigned int)f2bf(v.w) << 16);
            *(uint2*)&As[row * LDA + kq] = p;
        }
        // ---- stage B1/B3: 32x64 f32 -> bf16 LDS transposed [n][k]
#pragma unroll
        for (int j = 0; j < 2; ++j) {
            int f  = tid + 256 * j;                // float4 id (16 per row)
            int kr = f >> 4;
            int nq = (f & 15) << 2;
            float4 v1 = *(const float4*)(w1g + (size_t)(k0 + kr) * INTER + nq);
            float4 v3 = *(const float4*)(w3g + (size_t)(k0 + kr) * INTER + nq);
            B1s[(nq + 0) * LDA + kr] = f2bf(v1.x);
            B1s[(nq + 1) * LDA + kr] = f2bf(v1.y);
            B1s[(nq + 2) * LDA + kr] = f2bf(v1.z);
            B1s[(nq + 3) * LDA + kr] = f2bf(v1.w);
            B3s[(nq + 0) * LDA + kr] = f2bf(v3.x);
            B3s[(nq + 1) * LDA + kr] = f2bf(v3.y);
            B3s[(nq + 2) * LDA + kr] = f2bf(v3.z);
            B3s[(nq + 3) * LDA + kr] = f2bf(v3.w);
        }
        // wave-uniform prefetch of next K-step's dominant streams
        if (k0 + 32 < HIDDEN) {
            if (tid < 64) {
                __builtin_prefetch(w1g + (size_t)(k0 + 32 + (tid >> 1)) * INTER + ((tid & 1) << 5), 0, 1);
            } else if (tid < 128) {
                int u = tid - 64;
                __builtin_prefetch(w3g + (size_t)(k0 + 32 + (u >> 1)) * INTER + ((u & 1) << 5), 0, 1);
            } else {
                __builtin_prefetch(xg + (size_t)(tid - 128) * HIDDEN + k0 + 32, 0, 1);
            }
        }
        __syncthreads();

        // ---- fragments (A and B: K-runs {8h..8h+7,16+8h..})
        Frag a[2], b1[2], b3[2];
#pragma unroll
        for (int tm = 0; tm < 2; ++tm) {
            const unsigned short* p = &As[(32 * wm + 16 * tm + l16) * LDA + 8 * half];
            a[tm].q[0] = *(const uint4*)p;
            a[tm].q[1] = *(const uint4*)(p + 16);
        }
#pragma unroll
        for (int tn = 0; tn < 2; ++tn) {
            const unsigned short* p1 = &B1s[(32 * wn + 16 * tn + l16) * LDA + 8 * half];
            b1[tn].q[0] = *(const uint4*)p1;
            b1[tn].q[1] = *(const uint4*)(p1 + 16);
            const unsigned short* p3 = &B3s[(32 * wn + 16 * tn + l16) * LDA + 8 * half];
            b3[tn].q[0] = *(const uint4*)p3;
            b3[tn].q[1] = *(const uint4*)(p3 + 16);
        }
#pragma unroll
        for (int tm = 0; tm < 2; ++tm)
#pragma unroll
            for (int tn = 0; tn < 2; ++tn) {
                acc1[tm][tn] = __builtin_amdgcn_wmma_f32_16x16x32_bf16(
                    false, a[tm].v, false, b1[tn].v, (short)0, acc1[tm][tn], false, false);
                acc3[tm][tn] = __builtin_amdgcn_wmma_f32_16x16x32_bf16(
                    false, a[tm].v, false, b3[tn].v, (short)0, acc3[tm][tn], false, false);
            }
        __syncthreads();
    }

    // ---- SwiGLU epilogue -> bf16 activations
#pragma unroll
    for (int tm = 0; tm < 2; ++tm)
#pragma unroll
        for (int tn = 0; tn < 2; ++tn)
#pragma unroll
            for (int r = 0; r < 8; ++r) {
                int m = 32 * wm + 16 * tm + 8 * half + r;
                int n = n0 + 32 * wn + 16 * tn + l16;
                float h = acc1[tm][tn][r];
                float g = acc3[tm][tn][r];
                float s = h / (1.0f + __expf(-h));    // silu(h)
                act[(tokBase + m) * (size_t)INTER + n] = f2bf(s * g);
            }
}

// ---------------------------------------------------------------------------
// Kernel 2: out = act @ w2 (f32 accumulate, f32 out).
// Same tiling; A tile staged via GLOBAL_LOAD_ASYNC_TO_LDS_B128 (ASYNCcnt),
// which bypasses VGPRs entirely for the pure bf16 copy stream.
// ---------------------------------------------------------------------------
__global__ __launch_bounds__(256)
void moe_down(const unsigned short* __restrict__ act,
              const float* __restrict__ w2,
              float* __restrict__ out) {
    __shared__ unsigned short As[128 * LDA];
    __shared__ unsigned short Bs[ 64 * LDA];

    const int e    = blockIdx.z;
    const int m0   = blockIdx.y * 128;
    const int n0   = blockIdx.x * 64;
    const int tid  = threadIdx.x;
    const int lane = tid & 31;
    const int wid  = tid >> 5;
    const int wm   = wid & 3;
    const int wn   = wid >> 2;
    const int half = lane >> 4;
    const int l16  = lane & 15;

    const size_t tokBase = (size_t)e * TOK_PER_E + m0;
    const unsigned short* ag = act + tokBase * INTER;
    const float* w2g = w2 + (size_t)e * INTER * HIDDEN + n0;

    v8f acc[2][2] = {};

    for (int k0 = 0; k0 < INTER; k0 += 32) {
        // ---- stage A: 128x32 bf16, async direct-to-LDS (no VGPR roundtrip)
#pragma unroll
        for (int j = 0; j < 2; ++j) {
            int f   = tid + 256 * j;               // 16B chunk id (4 per row)
            int row = f >> 2;
            int kq  = (f & 3) << 3;
            const unsigned short* src = ag + (size_t)row * INTER + k0 + kq;
            unsigned dst = lds_off_u16(&As[row * LDA + kq]);
            asm volatile("global_load_async_to_lds_b128 %0, %1, off"
                         :: "v"(dst), "v"(src) : "memory");
        }
        // ---- stage B: w2 32x64 f32 -> bf16 LDS transposed [n][k]
#pragma unroll
        for (int j = 0; j < 2; ++j) {
            int f  = tid + 256 * j;
            int kr = f >> 4;
            int nq = (f & 15) << 2;
            float4 v = *(const float4*)(w2g + (size_t)(k0 + kr) * HIDDEN + nq);
            Bs[(nq + 0) * LDA + kr] = f2bf(v.x);
            Bs[(nq + 1) * LDA + kr] = f2bf(v.y);
            Bs[(nq + 2) * LDA + kr] = f2bf(v.z);
            Bs[(nq + 3) * LDA + kr] = f2bf(v.w);
        }
        if (k0 + 32 < INTER) {
            if (tid < 64) {
                __builtin_prefetch(w2g + (size_t)(k0 + 32 + (tid >> 1)) * HIDDEN + ((tid & 1) << 5), 0, 1);
            } else if (tid < 192) {
                __builtin_prefetch(ag + (size_t)(tid - 64) * INTER + k0 + 32, 0, 1);
            }
        }
        // Each wave drains its own async copies before signaling the barrier,
        // so after the barrier every wave sees the full A tile in LDS.
        wait_asynccnt0();
        __syncthreads();

        Frag a[2], b[2];
#pragma unroll
        for (int tm = 0; tm < 2; ++tm) {
            const unsigned short* p = &As[(32 * wm + 16 * tm + l16) * LDA + 8 * half];
            a[tm].q[0] = *(const uint4*)p;
            a[tm].q[1] = *(const uint4*)(p + 16);
        }
#pragma unroll
        for (int tn = 0; tn < 2; ++tn) {
            const unsigned short* p = &Bs[(32 * wn + 16 * tn + l16) * LDA + 8 * half];
            b[tn].q[0] = *(const uint4*)p;
            b[tn].q[1] = *(const uint4*)(p + 16);
        }
#pragma unroll
        for (int tm = 0; tm < 2; ++tm)
#pragma unroll
            for (int tn = 0; tn < 2; ++tn)
                acc[tm][tn] = __builtin_amdgcn_wmma_f32_16x16x32_bf16(
                    false, a[tm].v, false, b[tn].v, (short)0, acc[tm][tn], false, false);
        __syncthreads();
    }

    // ---- epilogue: f32 out
#pragma unroll
    for (int tm = 0; tm < 2; ++tm)
#pragma unroll
        for (int tn = 0; tn < 2; ++tn)
#pragma unroll
            for (int r = 0; r < 8; ++r) {
                int m = 32 * wm + 16 * tm + 8 * half + r;
                int n = n0 + 32 * wn + 16 * tn + l16;
                out[(tokBase + m) * (size_t)HIDDEN + n] = acc[tm][tn][r];
            }
}

// ---------------------------------------------------------------------------
extern "C" void kernel_launch(void* const* d_in, const int* in_sizes, int n_in,
                              void* d_out, int out_size, void* d_ws, size_t ws_size,
                              hipStream_t stream) {
    (void)in_sizes; (void)n_in; (void)out_size; (void)ws_size;
    const float* hiddens = (const float*)d_in[0];
    const float* w1      = (const float*)d_in[1];
    const float* w2      = (const float*)d_in[2];
    const float* w3      = (const float*)d_in[3];
    // d_in[4] = batch_sizes (int64): equal contiguous groups of 512 per the
    // problem setup; the grid is statically partitioned accordingly.

    unsigned short* act = (unsigned short*)d_ws;   // bf16 [4096][2816] = 23.1 MB
    float* out = (float*)d_out;

    dim3 blk(256);
    dim3 g1(INTER / 64, TOK_PER_E / 128, NUM_EXPERTS);   // 44 x 4 x 8
    hipLaunchKernelGGL(moe_up_gate, g1, blk, 0, stream, hiddens, w1, w3, act);

    dim3 g2(HIDDEN / 64, TOK_PER_E / 128, NUM_EXPERTS);  // 16 x 4 x 8
    hipLaunchKernelGGL(moe_down, g2, blk, 0, stream, act, w2, out);
}
